// DHSFNN_61254823575554
// MI455X (gfx1250) — matrix operand order repaired and checked by reading, not compiled, not run on hardware
//
#include <hip/hip_runtime.h>
#include <math.h>

// ---------------------------------------------------------------------------
// DH-SFNN forward, restructured as: big WMMA GEMMs (time-parallel) + cheap
// per-neuron scans (time-serial, neuron-parallel).
//   B=128 T=250 IN=700 H=512 OUT=20 BR=4, VTH=DT=1, WARMUP=10
// ---------------------------------------------------------------------------

typedef __attribute__((ext_vector_type(16))) _Float16 v16h;
typedef __attribute__((ext_vector_type(8)))  _Float16 v8h;
typedef __attribute__((ext_vector_type(8)))  float    v8f;

#define DEV __device__ __forceinline__

static constexpr int Bsz = 128;
static constexpr int T   = 250;
static constexpr int IN  = 700;
static constexpr int INP = 704;   // padded to multiple of 32 for WMMA K
static constexpr int H   = 512;
static constexpr int OUT = 20;
static constexpr int OUTP= 32;    // padded N for readout GEMM
static constexpr int BR  = 4;
static constexpr int HB  = H * BR;      // 2048
static constexpr int M   = Bsz * T;     // 32000 rows (multiple of 256)

DEV float sigmoidf_(float x) { return 1.0f / (1.0f + __expf(-x)); }

DEV v16h pack16(v8h lo, v8h hi) {
    return __builtin_shufflevector(lo, hi, 0, 1, 2, 3, 4, 5, 6, 7,
                                   8, 9, 10, 11, 12, 13, 14, 15);
}

// A-fragment (16x32 f16, row-major source): lanes 0-15 hold M=row, K {0..7,16..23};
// lanes 16-31 hold K {8..15,24..31}.
DEV v16h load_a_frag(const _Float16* __restrict__ A, int lda, int row0, int k0,
                     int lane) {
    const _Float16* p = A + (size_t)(row0 + (lane & 15)) * lda +
                        (k0 + ((lane >> 4) << 3));
    return pack16(*(const v8h*)p, *(const v8h*)(p + 16));
}

// B-fragment (32x16 f16) from row-major [N,K] weight: lane = column (Bw row),
// lanes 0-15 hold K {0..7,16..23}; lanes 16-31 hold K {8..15,24..31}.
DEV v16h load_b_frag(const _Float16* __restrict__ Bw, int ldb, int n0, int k0,
                     int lane) {
    const _Float16* p = Bw + (size_t)(n0 + (lane & 15)) * ldb +
                        (k0 + ((lane >> 4) << 3));
    return pack16(*(const v8h*)p, *(const v8h*)(p + 16));
}

// C/D layout: VGPR r -> M=r (lanes 0-15) or M=r+8 (lanes 16-31); N = lane&15.
DEV void store_c_frag(float* __restrict__ C, int ldc, int row0, int col0,
                      v8f acc, int lane) {
    int n  = col0 + (lane & 15);
    int rb = row0 + ((lane >> 4) << 3);
#pragma unroll
    for (int r = 0; r < 8; ++r) C[(size_t)(rb + r) * ldc + n] = acc[r];
}

// ---------------------------------------------------------------------------
// f16 operand prep (mask + pad + convert)
// ---------------------------------------------------------------------------
__global__ void prep_weight(const float* __restrict__ W, const float* __restrict__ mask,
                            _Float16* __restrict__ out, int Nreal, int Kreal,
                            int Npad, int Kpad) {
    int i = blockIdx.x * blockDim.x + threadIdx.x;
    if (i >= Npad * Kpad) return;
    int n = i / Kpad, k = i % Kpad;
    float v = 0.0f;
    if (n < Nreal && k < Kreal) {
        v = W[n * Kreal + k];
        if (mask) v *= mask[n * Kreal + k];
    }
    out[i] = (_Float16)v;
}

__global__ void prep_x(const float* __restrict__ x, _Float16* __restrict__ Xh,
                       int rows, int Kreal, int Kpad) {
    int i = blockIdx.x * blockDim.x + threadIdx.x;
    if (i >= rows * Kpad) return;
    int r = i / Kpad, k = i % Kpad;
    Xh[i] = (_Float16)(k < Kreal ? x[r * Kreal + k] : 0.0f);
}

// ---------------------------------------------------------------------------
// Full-tile WMMA GEMM: C[M,N] = A[M,K] * Bw[N,K]^T.
// Block = 128 threads = 4 waves in 2x2; wave tile 64x64 (4x4 fragments).
// Requires M % 128 == 0 and N % 128 == 0 -> no conditionals in the hot loop,
// EXEC stays all-1s around every WMMA.
// ---------------------------------------------------------------------------
__global__ __launch_bounds__(128) void wmma_gemm_full(
    const _Float16* __restrict__ A, int lda,
    const _Float16* __restrict__ Bw, int ldb,
    float* __restrict__ C, int ldc, int K) {
    const int lane = threadIdx.x & 31;
    const int wave = __builtin_amdgcn_readfirstlane(threadIdx.x >> 5);
    const int m0 = blockIdx.x * 128 + (wave >> 1) * 64;
    const int n0 = blockIdx.y * 128 + (wave & 1) * 64;

    v8f acc[4][4];
#pragma unroll
    for (int i = 0; i < 4; ++i)
#pragma unroll
        for (int j = 0; j < 4; ++j) acc[i][j] = {};

    for (int k0 = 0; k0 < K; k0 += 32) {
        v16h afr[4], bfr[4];
#pragma unroll
        for (int i = 0; i < 4; ++i) afr[i] = load_a_frag(A, lda, m0 + 16 * i, k0, lane);
#pragma unroll
        for (int j = 0; j < 4; ++j) bfr[j] = load_b_frag(Bw, ldb, n0 + 16 * j, k0, lane);
#pragma unroll
        for (int i = 0; i < 4; ++i)
#pragma unroll
            for (int j = 0; j < 4; ++j)
                acc[i][j] = __builtin_amdgcn_wmma_f32_16x16x32_f16(
                    false, afr[i], false, bfr[j], (short)0, acc[i][j], false, false);
    }

#pragma unroll
    for (int i = 0; i < 4; ++i)
#pragma unroll
        for (int j = 0; j < 4; ++j)
            store_c_frag(C, ldc, m0 + 16 * i, n0 + 16 * j, acc[i][j], lane);
}

// ---------------------------------------------------------------------------
// Narrow-N WMMA GEMM (N == 32, readout): 4 waves stacked along M, each wave
// computes a 64x32 tile (4x2 fragments). Requires M % 256 == 0. Branch-free.
// ---------------------------------------------------------------------------
__global__ __launch_bounds__(128) void wmma_gemm_n32(
    const _Float16* __restrict__ A, int lda,
    const _Float16* __restrict__ Bw, int ldb,
    float* __restrict__ C, int ldc, int K) {
    const int lane = threadIdx.x & 31;
    const int wave = __builtin_amdgcn_readfirstlane(threadIdx.x >> 5);
    const int m0 = blockIdx.x * 256 + wave * 64;

    v8f acc[4][2];
#pragma unroll
    for (int i = 0; i < 4; ++i)
#pragma unroll
        for (int j = 0; j < 2; ++j) acc[i][j] = {};

    for (int k0 = 0; k0 < K; k0 += 32) {
        v16h afr[4], bfr[2];
#pragma unroll
        for (int i = 0; i < 4; ++i) afr[i] = load_a_frag(A, lda, m0 + 16 * i, k0, lane);
#pragma unroll
        for (int j = 0; j < 2; ++j) bfr[j] = load_b_frag(Bw, ldb, 16 * j, k0, lane);
#pragma unroll
        for (int i = 0; i < 4; ++i)
#pragma unroll
            for (int j = 0; j < 2; ++j)
                acc[i][j] = __builtin_amdgcn_wmma_f32_16x16x32_f16(
                    false, afr[i], false, bfr[j], (short)0, acc[i][j], false, false);
    }

#pragma unroll
    for (int i = 0; i < 4; ++i)
#pragma unroll
        for (int j = 0; j < 2; ++j)
            store_c_frag(C, ldc, m0 + 16 * i, 16 * j, acc[i][j], lane);
}

// ---------------------------------------------------------------------------
// Per-neuron LIF scan over T. One thread per (b,h). Reads the precomputed
// branch currents [B*T, H*BR] (float4 per neuron per step, coalesced),
// writes f16 spikes [B*T, H].
// ---------------------------------------------------------------------------
__global__ void scan_layer(const float* __restrict__ Aact,   // [M, 2048]
                           const float* __restrict__ bias,   // [2048]
                           const float* __restrict__ tau_m,  // [512]
                           const float* __restrict__ tau_n,  // [2048]
                           _Float16* __restrict__ Sout) {    // [M, 512]
    int idx = blockIdx.x * blockDim.x + threadIdx.x;
    if (idx >= Bsz * H) return;
    int b = idx / H, h = idx % H;

    float alpha = sigmoidf_(tau_m[h]);
    float beta[BR], bb[BR];
#pragma unroll
    for (int j = 0; j < BR; ++j) {
        beta[j] = sigmoidf_(tau_n[h * BR + j]);
        bb[j]   = bias[h * BR + j];
    }
    float d[BR] = {0.f, 0.f, 0.f, 0.f};
    float m = 0.f, s = 0.f;

    for (int t = 0; t < T; ++t) {
        size_t row = (size_t)b * T + t;
        float4 a = *(const float4*)(Aact + row * HB + h * BR);
        d[0] = beta[0] * d[0] + (1.f - beta[0]) * (a.x + bb[0]);
        d[1] = beta[1] * d[1] + (1.f - beta[1]) * (a.y + bb[1]);
        d[2] = beta[2] * d[2] + (1.f - beta[2]) * (a.z + bb[2]);
        d[3] = beta[3] * d[3] + (1.f - beta[3]) * (a.w + bb[3]);
        float dsum = (d[0] + d[1]) + (d[2] + d[3]);
        m = m * alpha + (1.f - alpha) * dsum - s;       // DT=1, VTH=1
        s = (m - 1.0f) > 0.0f ? 1.0f : 0.0f;
        Sout[row * H + h] = (_Float16)s;
    }
}

// ---------------------------------------------------------------------------
// Readout: leaky integrator + softmax + sum over t >= WARMUP+1. One thread per
// batch element (trivial work: 128 * 250 * 20 flops).
// ---------------------------------------------------------------------------
__global__ void scan_readout(const float* __restrict__ A3,      // [M, 32]
                             const float* __restrict__ bo,      // [20]
                             const float* __restrict__ tau_mo,  // [20]
                             float* __restrict__ out) {         // [128, 20]
    int b = blockIdx.x * blockDim.x + threadIdx.x;
    if (b >= Bsz) return;
    float alpha[OUT], bia[OUT], mo[OUT], acc[OUT];
#pragma unroll
    for (int o = 0; o < OUT; ++o) {
        alpha[o] = sigmoidf_(tau_mo[o]);
        bia[o]   = bo[o];
        mo[o]    = 0.f;
        acc[o]   = 0.f;
    }
    for (int t = 0; t < T; ++t) {
        size_t row = (size_t)b * T + t;
        float mx = -1e30f;
#pragma unroll
        for (int o = 0; o < OUT; ++o) {
            float cur = A3[row * OUTP + o] + bia[o];
            mo[o] = mo[o] * alpha[o] + (1.f - alpha[o]) * cur;
            mx = fmaxf(mx, mo[o]);
        }
        if (t >= 11) {  // skip indices <= WARMUP, keep ys[WARMUP+1:]
            float e[OUT], sum = 0.f;
#pragma unroll
            for (int o = 0; o < OUT; ++o) { e[o] = __expf(mo[o] - mx); sum += e[o]; }
            float inv = 1.0f / sum;
#pragma unroll
            for (int o = 0; o < OUT; ++o) acc[o] += e[o] * inv;
        }
    }
#pragma unroll
    for (int o = 0; o < OUT; ++o) out[b * OUT + o] = acc[o];
}

// ---------------------------------------------------------------------------
// Host-side orchestration
// ---------------------------------------------------------------------------
extern "C" void kernel_launch(void* const* d_in, const int* in_sizes, int n_in,
                              void* d_out, int out_size, void* d_ws, size_t ws_size,
                              hipStream_t stream) {
    const float* x      = (const float*)d_in[0];
    const float* W1     = (const float*)d_in[1];
    const float* b1     = (const float*)d_in[2];
    const float* tau_m1 = (const float*)d_in[3];
    const float* tau_n1 = (const float*)d_in[4];
    const float* mask1  = (const float*)d_in[5];
    const float* W2     = (const float*)d_in[6];
    const float* b2     = (const float*)d_in[7];
    const float* tau_m2 = (const float*)d_in[8];
    const float* tau_n2 = (const float*)d_in[9];
    const float* mask2  = (const float*)d_in[10];
    const float* Wo     = (const float*)d_in[11];
    const float* bo     = (const float*)d_in[12];
    const float* tau_mo = (const float*)d_in[13];
    float* out = (float*)d_out;

    // carve workspace (offsets 256B aligned)
    char* ws = (char*)d_ws;
    size_t off = 0;
    auto carve = [&](size_t bytes) -> void* {
        void* p = ws + off;
        off = (off + bytes + 255) & ~(size_t)255;
        return p;
    };
    _Float16* Wm1h = (_Float16*)carve((size_t)HB * INP * 2);   //  2.9 MB
    _Float16* Wm2h = (_Float16*)carve((size_t)HB * H * 2);     //  2.1 MB
    _Float16* Woh  = (_Float16*)carve((size_t)OUTP * H * 2);   // 32 KB
    _Float16* Xh   = (_Float16*)carve((size_t)M * INP * 2);    // 45 MB
    _Float16* S1h  = (_Float16*)carve((size_t)M * H * 2);      // 33 MB
    _Float16* S2h  = (_Float16*)carve((size_t)M * H * 2);      // 33 MB
    float*    A3   = (float*)   carve((size_t)M * OUTP * 4);   //  4 MB
    float*    Abig = (float*)   carve((size_t)M * HB * 4);     // 262 MB (shared L1/L2 acts)
    (void)ws_size; (void)in_sizes; (void)n_in; (void)out_size;

    const int TPB = 256;
    // operand prep
    prep_weight<<<(HB * INP + TPB - 1) / TPB, TPB, 0, stream>>>(W1, mask1, Wm1h, HB, IN, HB, INP);
    prep_weight<<<(HB * H   + TPB - 1) / TPB, TPB, 0, stream>>>(W2, mask2, Wm2h, HB, H, HB, H);
    prep_weight<<<(OUTP * H + TPB - 1) / TPB, TPB, 0, stream>>>(Wo, nullptr, Woh, OUT, H, OUTP, H);
    prep_x<<<((size_t)M * INP + TPB - 1) / TPB, TPB, 0, stream>>>(x, Xh, M, IN, INP);

    // layer 1: X[32000,704] @ Wm1^T -> Abig[32000,2048]
    wmma_gemm_full<<<dim3(M / 128, HB / 128), 128, 0, stream>>>(
        Xh, INP, Wm1h, INP, Abig, HB, INP);
    scan_layer<<<(Bsz * H + TPB - 1) / TPB, TPB, 0, stream>>>(Abig, b1, tau_m1, tau_n1, S1h);

    // layer 2: S1[32000,512] @ Wm2^T -> Abig[32000,2048]
    wmma_gemm_full<<<dim3(M / 128, HB / 128), 128, 0, stream>>>(
        S1h, H, Wm2h, H, Abig, HB, H);
    scan_layer<<<(Bsz * H + TPB - 1) / TPB, TPB, 0, stream>>>(Abig, b2, tau_m2, tau_n2, S2h);

    // readout: S2[32000,512] @ Wo^T(padded to 32) -> A3[32000,32]
    wmma_gemm_n32<<<dim3(M / 256, 1), 128, 0, stream>>>(
        S2h, H, Woh, H, A3, OUTP, H);
    scan_readout<<<1, 128, 0, stream>>>(A3, bo, tau_mo, out);
}
